// MultiHeadAttention_28570122453978
// MI455X (gfx1250) — matrix operand order repaired and checked by reading, not compiled, not run on hardware
//
#include <hip/hip_runtime.h>
#include <hip/hip_bf16.h>

#define D_MODEL 1024
#define HEADS 16
#define HEAD_DIM 64
#define BATCH 2
#define SEQ 2048
#define ROWS (BATCH * SEQ) /* 4096 */

typedef __attribute__((ext_vector_type(16))) _Float16 v16h;
typedef __attribute__((ext_vector_type(8)))  _Float16 v8hh;
typedef __attribute__((ext_vector_type(8)))  float    v8f;
typedef __attribute__((ext_vector_type(4)))  float    v4ff;

// ---------------------------------------------------------------------------
// WMMA helpers (gfx1250, wave32). Layouts per CDNA5 ISA 7.12.2:
//  A 16x32 f16 : lane = M (mod 16); lane-half selects K quarters {h*8.., 16+h*8..}
//  B 32x16 f16 : lane = N (mod 16); lane-half selects K quarters {h*8.., 16+h*8..}
//  C/D 16x16 f32: lane = N (mod 16); VGPR r holds row r (lanes 0-15) / 8+r (16-31)
// ---------------------------------------------------------------------------
__device__ __forceinline__ v8f wmma_f16(v16h a, v16h b, v8f c) {
    return __builtin_amdgcn_wmma_f32_16x16x32_f16(
        false, a, false, b, (short)0, c, false, false);
}

__device__ __forceinline__ v16h load_a16(const _Float16* __restrict__ base,
                                         int ld, int m0, int k0) {
    const int lane = threadIdx.x & 31;
    const int m = lane & 15;
    const int h = lane >> 4;
    const _Float16* p = base + (size_t)(m0 + m) * ld + k0 + h * 8;
    v8hh lo = *(const v8hh*)p;          // K = k0 + h*8 .. +7
    v8hh hi = *(const v8hh*)(p + 16);   // K = k0 + 16 + h*8 .. +7
    v16h a;
#pragma unroll
    for (int i = 0; i < 8; ++i) { a[i] = lo[i]; a[i + 8] = hi[i]; }
    return a;
}

// B-fragment where memory holds B^T row-major: B(k,n) = base[n*ld + k]
__device__ __forceinline__ v16h load_bt16(const _Float16* __restrict__ base,
                                          int ld, int k0, int n0) {
    const int lane = threadIdx.x & 31;
    const int n = lane & 15;
    const int h = lane >> 4;
    const _Float16* p = base + (size_t)(n0 + n) * ld + k0 + h * 8;
    v8hh lo = *(const v8hh*)p;
    v8hh hi = *(const v8hh*)(p + 16);
    v16h b;
#pragma unroll
    for (int i = 0; i < 8; ++i) { b[i] = lo[i]; b[i + 8] = hi[i]; }
    return b;
}

// ---- CDNA5 async global->LDS (16B per lane per op), tracked by ASYNCcnt ----
__device__ __forceinline__ void async_ld16(const void* g, void* l) {
    asm volatile("global_load_async_to_lds_b128 %0, %1, off"
                 :: "v"((unsigned)(uintptr_t)l),
                    "v"((unsigned long long)(uintptr_t)g)
                 : "memory");
}
__device__ __forceinline__ void wait_async0() {
    asm volatile("s_wait_asynccnt 0x0" ::: "memory");
}

// ---- 16-lane row-max butterfly on the VALU DPP pipe (no LDS, no waits) ----
// xor-masks {1,2,7,15} are linearly independent -> full 16-lane reduction.
template <int CTRL>
__device__ __forceinline__ void dpp_max_step(float (&v)[8]) {
    float t[8];
#pragma unroll
    for (int r = 0; r < 8; ++r)
        t[r] = __int_as_float(__builtin_amdgcn_update_dpp(
            0, __float_as_int(v[r]), CTRL, 0xF, 0xF, true));
#pragma unroll
    for (int r = 0; r < 8; ++r) v[r] = fmaxf(v[r], t[r]);
}
__device__ __forceinline__ void halfmax16x8(float (&v)[8]) {
    dpp_max_step<0xB1>(v);   // quad_perm(1,0,3,2) : xor 1
    dpp_max_step<0x4E>(v);   // quad_perm(2,3,0,1) : xor 2
    dpp_max_step<0x141>(v);  // row_half_mirror    : xor 7
    dpp_max_step<0x140>(v);  // row_mirror         : xor 15
}

// ---------------------------------------------------------------------------
// f32 -> f16 conversion
// ---------------------------------------------------------------------------
__global__ void cvt_f32_to_f16_kernel(const float* __restrict__ src,
                                      _Float16* __restrict__ dst, int n8) {
    int i = blockIdx.x * blockDim.x + threadIdx.x;
    const int stride = gridDim.x * blockDim.x;
    for (; i < n8; i += stride) {
        const v4ff a = ((const v4ff*)src)[2 * i + 0];
        const v4ff b = ((const v4ff*)src)[2 * i + 1];
        v8hh h;
#pragma unroll
        for (int j = 0; j < 4; ++j) { h[j] = (_Float16)a[j]; h[j + 4] = (_Float16)b[j]; }
        ((v8hh*)dst)[i] = h;
    }
}

// ---------------------------------------------------------------------------
// Projection GEMM: Y[m,n] = (sum_k X[m,k]*W[n,k] + bias[n]) * scale
// One wave -> 32(M) x 64(N): per k-step 2 A-frags + 4 B-frags -> 8 v_wmma
// MODE 0: f16 [B,H,S,Dh] | MODE 1: f16 [B,H,Dh,S] | MODE 2: f32 [ROWS,D_MODEL]
// scale=0.125 for the Q projection folds 1/sqrt(Dh) out of the attention loop.
// ---------------------------------------------------------------------------
template <int MODE>
__global__ void proj32x64_kernel(const _Float16* __restrict__ X,
                                 const _Float16* __restrict__ W,
                                 const float* __restrict__ bias,
                                 void* __restrict__ Y, float scale) {
    const int m0 = blockIdx.x * 32;
    const int n0 = blockIdx.y * 64;

    v8f acc[2][4] = {};
    for (int k0 = 0; k0 < D_MODEL; k0 += 32) {
        const v16h a0 = load_a16(X, D_MODEL, m0,      k0);
        const v16h a1 = load_a16(X, D_MODEL, m0 + 16, k0);
#pragma unroll
        for (int t = 0; t < 4; ++t) {
            const v16h b = load_bt16(W, D_MODEL, k0, n0 + 16 * t);
            acc[0][t] = wmma_f16(a0, b, acc[0][t]);
            acc[1][t] = wmma_f16(a1, b, acc[1][t]);
        }
    }

    const int lane  = threadIdx.x & 31;
    const int nlane = lane & 15;
    const int mh8   = (lane >> 4) * 8;
#pragma unroll
    for (int u = 0; u < 2; ++u)
#pragma unroll
    for (int t = 0; t < 4; ++t) {
        const int n = n0 + 16 * t + nlane;
        const float bval = bias[n];
#pragma unroll
        for (int r = 0; r < 8; ++r) {
            const int m = m0 + 16 * u + mh8 + r;
            const float y = (acc[u][t][r] + bval) * scale;
            if (MODE == 2) {
                ((float*)Y)[(size_t)m * D_MODEL + n] = y;
            } else {
                const int b = m / SEQ, s = m % SEQ;
                const int h = n / HEAD_DIM, d = n % HEAD_DIM;
                if (MODE == 0)
                    ((_Float16*)Y)[((size_t)(b * HEADS + h) * SEQ + s) * HEAD_DIM + d] = (_Float16)y;
                else
                    ((_Float16*)Y)[((size_t)(b * HEADS + h) * HEAD_DIM + d) * SEQ + s] = (_Float16)y;
            }
        }
    }
}

// ---------------------------------------------------------------------------
// Flash attention, one wave per (b,h,16-row q-tile). KV walked 32 wide.
// K/V tiles double-buffered in LDS via global_load_async_to_lds_b128.
// Row-max via DPP butterfly; row-sums of P via WMMA against an all-ones B.
// Q already carries the 1/sqrt(Dh) scale from its projection.
// ---------------------------------------------------------------------------
__global__ void attn_kernel(const _Float16* __restrict__ Qp,
                            const _Float16* __restrict__ Kp,
                            const _Float16* __restrict__ Vt,
                            _Float16* __restrict__ Ctx) {
    __shared__ __align__(32) _Float16 Kbuf[2][32 * HEAD_DIM]; // 2 x 4KB [kv][d]
    __shared__ __align__(32) _Float16 Vbuf[2][HEAD_DIM * 32]; // 2 x 4KB [d][kv]
    __shared__ __align__(32) _Float16 Plds[16 * 32];          // 1KB

    const int lane  = threadIdx.x & 31;
    const int nlane = lane & 15;
    const int mh8   = (lane >> 4) * 8;

    const int q0 = blockIdx.x * 16;
    const int bh = blockIdx.y;
    const _Float16* Qb = Qp + (size_t)bh * SEQ * HEAD_DIM;
    const char*     KbG = (const char*)(Kp + (size_t)bh * SEQ * HEAD_DIM);
    const char*     VbG = (const char*)(Vt + (size_t)bh * HEAD_DIM * SEQ);

    const v16h aq0 = load_a16(Qb, HEAD_DIM, q0, 0);
    const v16h aq1 = load_a16(Qb, HEAD_DIM, q0, 32);

    v16h ones;
#pragma unroll
    for (int i = 0; i < 16; ++i) ones[i] = (_Float16)1.0f;

    // K tile (32 rows x 64 d) is 4KB CONTIGUOUS in [s,d]; V^T tile is 64 rows x 64B
    auto issueK = [&](int buf, int kv0) {
#pragma unroll
        for (int c = 0; c < 8; ++c) {
            const int j = c * 32 + lane;                      // 16B chunk id
            async_ld16(KbG + (size_t)kv0 * 128 + (size_t)j * 16, &Kbuf[buf][0] + j * 8);
        }
    };
    auto issueV = [&](int buf, int kv0) {
#pragma unroll
        for (int c = 0; c < 8; ++c) {
            const int j = c * 32 + lane;
            const int row = j >> 2, col = j & 3;              // row=d, 4x16B per row
            async_ld16(VbG + (size_t)row * (SEQ * 2) + (size_t)kv0 * 2 + col * 16,
                       &Vbuf[buf][0] + j * 8);
        }
    };

    float mrow[8], lrow[8];
#pragma unroll
    for (int r = 0; r < 8; ++r) { mrow[r] = -1e30f; lrow[r] = 0.0f; }
    v8f o[4] = {};

    issueK(0, 0);
    issueV(0, 0);

    int buf = 0;
    for (int kv0 = 0; kv0 < SEQ; kv0 += 32, buf ^= 1) {
        wait_async0();
        __syncthreads();
        if (kv0 + 32 < SEQ) { issueK(buf ^ 1, kv0 + 32); issueV(buf ^ 1, kv0 + 32); }

        const _Float16* Kt = &Kbuf[buf][0];
        const _Float16* Vv = &Vbuf[buf][0];

        // ---- scores (already scaled): two 16-wide kv sub-tiles, 4 WMMAs --
        v8f sc[2];
#pragma unroll
        for (int c = 0; c < 2; ++c) {
            v8f z = {};
            z = wmma_f16(aq0, load_bt16(Kt, HEAD_DIM, 0,  16 * c), z);
            z = wmma_f16(aq1, load_bt16(Kt, HEAD_DIM, 32, 16 * c), z);
            sc[c] = z;
        }

        // ---- online softmax: DPP row max, exp in place -------------------
        float vmax[8];
#pragma unroll
        for (int r = 0; r < 8; ++r) vmax[r] = fmaxf(sc[0][r], sc[1][r]);
        halfmax16x8(vmax);

        float alpha[8];
#pragma unroll
        for (int r = 0; r < 8; ++r) {
            const float mnew = fmaxf(mrow[r], vmax[r]);
            alpha[r] = __expf(mrow[r] - mnew);
            sc[0][r] = __expf(sc[0][r] - mnew);
            sc[1][r] = __expf(sc[1][r] - mnew);
            mrow[r]  = mnew;
        }
#pragma unroll
        for (int t = 0; t < 4; ++t)
#pragma unroll
            for (int r = 0; r < 8; ++r) o[t][r] *= alpha[r];

        // ---- P: D-layout -> LDS (16x32) -> A-layout ---------------------
#pragma unroll
        for (int c = 0; c < 2; ++c)
#pragma unroll
            for (int r = 0; r < 8; ++r)
                Plds[(mh8 + r) * 32 + 16 * c + nlane] = (_Float16)sc[c][r];
        __syncthreads();

        const _Float16* pr = &Plds[nlane * 32 + (lane >> 4) * 8];
        v8hh lo = *(const v8hh*)pr;
        v8hh hi = *(const v8hh*)(pr + 16);
        v16h ap;
#pragma unroll
        for (int i = 0; i < 8; ++i) { ap[i] = lo[i]; ap[i + 8] = hi[i]; }

        // ---- row-sum of P via WMMA against ones(32x16) ------------------
        {
            v8f zz = {};
            const v8f ps = wmma_f16(ap, ones, zz);
#pragma unroll
            for (int r = 0; r < 8; ++r) lrow[r] = lrow[r] * alpha[r] + ps[r];
        }

        // ---- O += P @ V (V^T tile in LDS -> contiguous B-frags) ---------
#pragma unroll
        for (int t = 0; t < 4; ++t)
            o[t] = wmma_f16(ap, load_bt16(Vv, 32, 0, 16 * t), o[t]);
    }

    // ---- normalize, write ctx as [B,S,H*Dh] f16 ------------------------
    const int b = bh / HEADS, h = bh % HEADS;
#pragma unroll
    for (int t = 0; t < 4; ++t)
#pragma unroll
        for (int r = 0; r < 8; ++r) {
            const float y = o[t][r] / lrow[r];
            const int s   = q0 + mh8 + r;
            const int col = h * HEAD_DIM + t * 16 + nlane;
            Ctx[(size_t)(b * SEQ + s) * D_MODEL + col] = (_Float16)y;
        }
}

// ---------------------------------------------------------------------------
extern "C" void kernel_launch(void* const* d_in, const int* in_sizes, int n_in,
                              void* d_out, int out_size, void* d_ws, size_t ws_size,
                              hipStream_t stream) {
    (void)in_sizes; (void)n_in; (void)out_size; (void)ws_size;

    const float* Q  = (const float*)d_in[0];
    const float* K  = (const float*)d_in[1];
    const float* V  = (const float*)d_in[2];
    const float* Wq = (const float*)d_in[3];
    const float* bq = (const float*)d_in[4];
    const float* Wk = (const float*)d_in[5];
    const float* bk = (const float*)d_in[6];
    const float* Wv = (const float*)d_in[7];
    const float* bv = (const float*)d_in[8];
    const float* Wo = (const float*)d_in[9];
    const float* bo = (const float*)d_in[10];

    const size_t MN = (size_t)ROWS * D_MODEL;
    const size_t WW = (size_t)D_MODEL * D_MODEL;

    char* w = (char*)d_ws;
    _Float16* Q16  = (_Float16*)w; w += MN * 2;
    _Float16* K16  = (_Float16*)w; w += MN * 2;
    _Float16* V16  = (_Float16*)w; w += MN * 2;
    _Float16* Wq16 = (_Float16*)w; w += WW * 2;
    _Float16* Wk16 = (_Float16*)w; w += WW * 2;
    _Float16* Wv16 = (_Float16*)w; w += WW * 2;
    _Float16* Wo16 = (_Float16*)w; w += WW * 2;
    _Float16* Qp   = (_Float16*)w; w += MN * 2;  // [B,H,S,Dh], pre-scaled 1/8
    _Float16* Kp   = (_Float16*)w; w += MN * 2;  // [B,H,S,Dh]
    _Float16* Vtp  = (_Float16*)w; w += MN * 2;  // [B,H,Dh,S]
    _Float16* Ctx  = (_Float16*)w; w += MN * 2;  // [B,S,H*Dh]

    const int cvtBlk = 256, cvtGrid = 1024;
    cvt_f32_to_f16_kernel<<<cvtGrid, cvtBlk, 0, stream>>>(Q,  Q16,  (int)(MN / 8));
    cvt_f32_to_f16_kernel<<<cvtGrid, cvtBlk, 0, stream>>>(K,  K16,  (int)(MN / 8));
    cvt_f32_to_f16_kernel<<<cvtGrid, cvtBlk, 0, stream>>>(V,  V16,  (int)(MN / 8));
    cvt_f32_to_f16_kernel<<<cvtGrid, cvtBlk, 0, stream>>>(Wq, Wq16, (int)(WW / 8));
    cvt_f32_to_f16_kernel<<<cvtGrid, cvtBlk, 0, stream>>>(Wk, Wk16, (int)(WW / 8));
    cvt_f32_to_f16_kernel<<<cvtGrid, cvtBlk, 0, stream>>>(Wv, Wv16, (int)(WW / 8));
    cvt_f32_to_f16_kernel<<<cvtGrid, cvtBlk, 0, stream>>>(Wo, Wo16, (int)(WW / 8));

    dim3 pgrid(ROWS / 32, D_MODEL / 64);
    proj32x64_kernel<0><<<pgrid, 32, 0, stream>>>(Q16, Wq16, bq, (void*)Qp, 0.125f);
    proj32x64_kernel<0><<<pgrid, 32, 0, stream>>>(K16, Wk16, bk, (void*)Kp, 1.0f);
    proj32x64_kernel<1><<<pgrid, 32, 0, stream>>>(V16, Wv16, bv, (void*)Vtp, 1.0f);

    dim3 agrid(SEQ / 16, BATCH * HEADS);
    attn_kernel<<<agrid, 32, 0, stream>>>(Qp, Kp, Vtp, Ctx);

    proj32x64_kernel<2><<<pgrid, 32, 0, stream>>>(Ctx, Wo16, bo, d_out, 1.0f);
}
